// HoppingBlock_25426206392411
// MI455X (gfx1250) — matrix operand order repaired and checked by reading, not compiled, not run on hardware
//
#include <hip/hip_runtime.h>
#include <hip/hip_bf16.h>
#include <math.h>

// ---------------------------------------------------------------------------
// Problem constants (from reference)
// ---------------------------------------------------------------------------
#define N_NODES 50000
#define N_EDGES 100000   // line-graph nodes (original edges)
#define N_LE    400000   // line-graph edges
#define HEADS   4
#define HD      256
#define D64     64

// ---------------------------------------------------------------------------
// CDNA5 WMMA types (probe-confirmed signatures)
// ---------------------------------------------------------------------------
typedef __attribute__((ext_vector_type(16))) __bf16 v16bf;
typedef __attribute__((ext_vector_type(8)))  __bf16 v8bf;
typedef __attribute__((ext_vector_type(8)))  float  v8f;

#if defined(__has_builtin)
#if __has_builtin(__builtin_amdgcn_cvt_pk_bf16_f32)
#define HAVE_CVT_PK_BF16 1
#endif
#endif

// f32 -> bf16 (round-half-up: matches RNE except on exact 0x8000 tails).
// Single value: v_add + v_lshr.  Pair: 2x v_add + 1x v_perm_b32.
__device__ __forceinline__ unsigned short bf1(float f) {
    return (unsigned short)((__float_as_uint(f) + 0x8000u) >> 16);
}
__device__ __forceinline__ unsigned pack_bf2(float lo, float hi) {
#ifdef HAVE_CVT_PK_BF16
    auto v = __builtin_amdgcn_cvt_pk_bf16_f32(lo, hi);   // single VOP pack
    unsigned r;
    __builtin_memcpy(&r, &v, sizeof(r));
    return r;
#else
    unsigned ulo = __float_as_uint(lo) + 0x8000u;
    unsigned uhi = __float_as_uint(hi) + 0x8000u;
    // v_perm_b32: out = {uhi[31:24], uhi[23:16], ulo[31:24], ulo[23:16]}
    return __builtin_amdgcn_perm(uhi, ulo, 0x07060302u);
#endif
}

// order-preserving float <-> uint mapping for atomicMax-based segment max
__device__ __forceinline__ unsigned f2ord(float f) {
    unsigned u = __float_as_uint(f);
    return (u & 0x80000000u) ? ~u : (u | 0x80000000u);
}
__device__ __forceinline__ float ord2f(unsigned u) {
    return (u & 0x80000000u) ? __uint_as_float(u & 0x7FFFFFFFu)
                             : __uint_as_float(~u);
}
#define NEG_INF_ORD 0x007FFFFFu   // f2ord(-inf)

// ---------------------------------------------------------------------------
// WMMA GEMM:  C[M,N] = act(A[M,K] @ W[K,N] + bias[N])
//   - fp32 in HBM; bf16 on the LDS stage; f32 WMMA accumulate
//   - block = 256 threads = 8 waves (wave32); tile 128(M) x 64(N), K-step 32
//   - DOUBLE-BUFFERED LDS (30.7KB of 320KB WGP pool) with register staging:
//     next tile's global_load_b128s issue before the current tile's WMMAs,
//     one barrier per K-step
//   - templated on FULL: interior blocks (all but the last grid.x block)
//     take a branch-free, guard-free staging loop and epilogue
//   - LDS rows padded to 40 bf16 (80B): 16-lane b128 fragment reads are
//     bank-conflict-free (bank stride 20) and A staging stores are
//     8B-aligned ds_store_b64
// ---------------------------------------------------------------------------
#define BM 128
#define BN 64
#define BK 32
#define LDP 40   // padded LDS row pitch in bf16 elements (80 bytes)

template <bool FULL>
__device__ __forceinline__ void gemm_body(
    const float* __restrict__ A, const float* __restrict__ W,
    const float* __restrict__ bias, float* __restrict__ C,
    int M, int K, int N, int act,
    unsigned short* AsU, unsigned short* BsU, int rowBase, int colBase)
{
    const int tid  = threadIdx.x;
    const int wid  = tid >> 5;        // 0..7 -> 16-row strip of the 128-row tile
    const int lane = tid & 31;
    const int half = lane >> 4;       // lane 0-15 vs 16-31 (ISA layout halves)
    const int lr   = lane & 15;

    // per-thread staging coordinates (constant across chunks)
    const int ar0 = tid >> 3;         // A row base (rows ar0, ar0+32, +64, +96)
    const int ac4 = (tid & 7) * 4;    // A float4 column offset
    const int bk0 = tid >> 4;         // B k base (kk = bk0, bk0+16)
    const int bn4 = (tid & 15) * 4;   // B col offset (4 cols)

    float4 areg[4];
    float4 breg[2];

    auto load_tile = [&](int k0) {
        #pragma unroll
        for (int i = 0; i < 4; ++i) {
            int gr = rowBase + ar0 + 32 * i;
            if (FULL) {
                areg[i] = *(const float4*)(A + (size_t)gr * K + (k0 + ac4));
            } else {
                int grc = (gr < M) ? gr : 0;       // clamp, then select-zero
                float4 t = *(const float4*)(A + (size_t)grc * K + (k0 + ac4));
                bool ok = (gr < M);
                areg[i].x = ok ? t.x : 0.0f; areg[i].y = ok ? t.y : 0.0f;
                areg[i].z = ok ? t.z : 0.0f; areg[i].w = ok ? t.w : 0.0f;
            }
        }
        #pragma unroll
        for (int i = 0; i < 2; ++i)
            breg[i] = *(const float4*)(W + (size_t)(k0 + bk0 + 16 * i) * N +
                                       (colBase + bn4));
    };

    auto store_tile = [&](int buf) {
        unsigned short* Ab = AsU + buf * (BM * LDP);
        unsigned short* Bb = BsU + buf * (BN * LDP);
        #pragma unroll
        for (int i = 0; i < 4; ++i) {
            uint2 p;
            p.x = pack_bf2(areg[i].x, areg[i].y);
            p.y = pack_bf2(areg[i].z, areg[i].w);
            *(uint2*)(Ab + (ar0 + 32 * i) * LDP + ac4) = p;   // ds_store_b64
        }
        #pragma unroll
        for (int i = 0; i < 2; ++i) {
            int kk = bk0 + 16 * i;
            Bb[(bn4 + 0) * LDP + kk] = bf1(breg[i].x);
            Bb[(bn4 + 1) * LDP + kk] = bf1(breg[i].y);
            Bb[(bn4 + 2) * LDP + kk] = bf1(breg[i].z);
            Bb[(bn4 + 3) * LDP + kk] = bf1(breg[i].w);
        }
    };

    v8f acc[4];
    #pragma unroll
    for (int f = 0; f < 4; ++f)
        #pragma unroll
        for (int e = 0; e < 8; ++e) acc[f][e] = 0.0f;

    const int T = K / BK;
    load_tile(0);
    store_tile(0);
    __syncthreads();

    for (int t = 0; t < T; ++t) {
        // issue next tile's global loads before the math so their latency
        // hides behind the ds_load+WMMA block
        if (t + 1 < T) load_tile((t + 1) * BK);
        if (t + 2 < T) {   // L2 prefetch two tiles ahead (global_prefetch_b8)
            int k2 = (t + 2) * BK;
            int gr = rowBase + ar0;
            if (FULL || gr < M)
                __builtin_prefetch(A + (size_t)gr * K + (k2 + ac4), 0, 1);
            __builtin_prefetch(W + (size_t)(k2 + bk0) * N + (colBase + bn4), 0, 1);
        }

        {
            const int buf = t & 1;
            const __bf16* Ab = (const __bf16*)(AsU + buf * (BM * LDP));
            const __bf16* Bb = (const __bf16*)(BsU + buf * (BN * LDP));

            // A fragment (16x32 bf16, ISA §7.12.2):
            //  half 0: elems 0-7 -> K=0..7,  8-15 -> K=16..23
            //  half 1: elems 0-7 -> K=8..15, 8-15 -> K=24..31
            const __bf16* ap = Ab + (wid * 16 + lr) * LDP;
            const int kb = half * 8;
            v8bf alo = *(const v8bf*)(ap + kb);
            v8bf ahi = *(const v8bf*)(ap + 16 + kb);
            v16bf afrag = __builtin_shufflevector(alo, ahi,
                0,1,2,3,4,5,6,7,8,9,10,11,12,13,14,15);

            const int kh = half * 8;    // same K layout as A fragment
            #pragma unroll
            for (int f = 0; f < 4; ++f) {
                const __bf16* bp = Bb + (f * 16 + lr) * LDP + kh;
                v8bf blo = *(const v8bf*)(bp);
                v8bf bhi = *(const v8bf*)(bp + 16);
                v16bf bfrag = __builtin_shufflevector(blo, bhi,
                    0,1,2,3,4,5,6,7,8,9,10,11,12,13,14,15);
                acc[f] = __builtin_amdgcn_wmma_f32_16x16x32_bf16(
                    false, afrag, false, bfrag, (short)0, acc[f], false, false);
            }
        }

        if (t + 1 < T) {
            store_tile((t + 1) & 1);   // other buffer: no conflict with reads
            __syncthreads();           // one barrier per K-step
        }
    }

    // ---- epilogue: C/D layout (8 VGPRs, row = e + 8*half, col = lane&15) ----
    #pragma unroll
    for (int f = 0; f < 4; ++f) {
        int col = colBase + f * 16 + lr;
        float b = bias ? bias[col] : 0.0f;
        #pragma unroll
        for (int e = 0; e < 8; ++e) {
            int row = rowBase + wid * 16 + half * 8 + e;
            if (FULL || row < M) {
                float v = acc[f][e] + b;
                if (act == 1) v = (v > 0.0f) ? v : 0.01f * v;  // leaky 0.01
                C[(size_t)row * N + col] = v;
            }
        }
    }
}

__global__ __launch_bounds__(256)
void wmma_gemm_kernel(const float* __restrict__ A, const float* __restrict__ W,
                      const float* __restrict__ bias, float* __restrict__ C,
                      int M, int K, int N, int act)
{
    __shared__ __align__(16) unsigned short AsU[2 * BM * LDP];  // double-buffered
    __shared__ __align__(16) unsigned short BsU[2 * BN * LDP];
    const int rowBase = blockIdx.x * BM;
    const int colBase = blockIdx.y * BN;
    if (rowBase + BM <= M)   // uniform per-block branch: interior fast path
        gemm_body<true >(A, W, bias, C, M, K, N, act, AsU, BsU, rowBase, colBase);
    else
        gemm_body<false>(A, W, bias, C, M, K, N, act, AsU, BsU, rowBase, colBase);
}

// ---------------------------------------------------------------------------
// Elementwise / edge (scatter-gather) kernels — f32, wave-per-edge layout
// ---------------------------------------------------------------------------
__global__ void init_u32_kernel(unsigned* p, unsigned v, int n) {
    int i = blockIdx.x * blockDim.x + threadIdx.x;
    if (i < n) p[i] = v;
}

__global__ __launch_bounds__(256)
void gine_edge_kernel(const float* __restrict__ h0, const float* __restrict__ pxg,
                      const int* __restrict__ esrc, const int* __restrict__ edst,
                      const int* __restrict__ enode, float* __restrict__ agg)
{
    int le = blockIdx.x * 8 + (threadIdx.x >> 5);
    if (le >= N_LE) return;
    int lane = threadIdx.x & 31;
    int src = esrc[le], dst = edst[le], node = enode[le];
    #pragma unroll
    for (int j = 0; j < 2; ++j) {
        int d = lane + 32 * j;
        float m = h0[(size_t)src * D64 + d] + pxg[(size_t)node * D64 + d];
        m = m > 0.0f ? m : 0.0f;
        atomicAdd(&agg[(size_t)dst * D64 + d], m);
    }
}

__global__ void gine_combine_kernel(const float* h0, const float* agg,
                                    const float* eps, float* g) {
    int i = blockIdx.x * blockDim.x + threadIdx.x;
    if (i < N_EDGES * D64) g[i] = (1.0f + eps[0]) * h0[i] + agg[i];
}

__global__ __launch_bounds__(256)
void tc_score_kernel(const float* __restrict__ q, const float* __restrict__ k,
                     const float* __restrict__ e, const int* __restrict__ esrc,
                     const int* __restrict__ edst, const int* __restrict__ enode,
                     float* __restrict__ score, unsigned* __restrict__ mkey)
{
    int le = blockIdx.x * 8 + (threadIdx.x >> 5);
    if (le >= N_LE) return;
    int lane = threadIdx.x & 31;
    int src = esrc[le], dst = edst[le], node = enode[le];
    const float* qd = q + (size_t)dst * HD;
    const float* ks = k + (size_t)src * HD;
    const float* en = e + (size_t)node * HD;
    float acc[HEADS] = {0.f, 0.f, 0.f, 0.f};
    #pragma unroll
    for (int j = 0; j < 8; ++j) {          // head = (lane + 32j) >> 6 = j >> 1
        int d = lane + 32 * j;
        acc[j >> 1] += qd[d] * (ks[d] + en[d]);
    }
    #pragma unroll
    for (int h = 0; h < HEADS; ++h) {
        float s = acc[h];
        #pragma unroll
        for (int off = 16; off > 0; off >>= 1) s += __shfl_xor(s, off);
        if (lane == 0) {
            s *= 0.125f;                   // 1/sqrt(64)
            score[(size_t)le * HEADS + h] = s;
            atomicMax(&mkey[(size_t)dst * HEADS + h], f2ord(s));
        }
    }
}

__global__ void tc_expsum_kernel(const float* score, const unsigned* mkey,
                                 const int* edst, float* exb, float* ssum) {
    int i = blockIdx.x * blockDim.x + threadIdx.x;
    if (i >= N_LE * HEADS) return;
    int le = i >> 2, h = i & 3;
    int dst = edst[le];
    float ex = expf(score[i] - ord2f(mkey[(size_t)dst * HEADS + h]));
    exb[i] = ex;
    atomicAdd(&ssum[(size_t)dst * HEADS + h], ex);
}

__global__ __launch_bounds__(256)
void tc_out_kernel(const float* __restrict__ v, const float* __restrict__ e,
                   const int* __restrict__ esrc, const int* __restrict__ edst,
                   const int* __restrict__ enode, const float* __restrict__ exb,
                   const float* __restrict__ ssum, float* __restrict__ out)
{
    int le = blockIdx.x * 8 + (threadIdx.x >> 5);
    if (le >= N_LE) return;
    int lane = threadIdx.x & 31;
    int src = esrc[le], dst = edst[le], node = enode[le];
    #pragma unroll
    for (int j = 0; j < 8; ++j) {
        int d = lane + 32 * j;
        int h = j >> 1;
        float alpha = exb[(size_t)le * HEADS + h] /
                      (ssum[(size_t)dst * HEADS + h] + 1e-16f);
        float val = (v[(size_t)src * HD + d] + e[(size_t)node * HD + d]) * alpha;
        atomicAdd(&out[(size_t)dst * HD + d], val);
    }
}

__global__ void add_kernel(const float* a, const float* b, float* c, int n) {
    int i = blockIdx.x * blockDim.x + threadIdx.x;
    if (i < n) c[i] = a[i] + b[i];
}

__global__ __launch_bounds__(256)
void gat_score_kernel(const float* __restrict__ xl, const float* __restrict__ xr,
                      const float* __restrict__ eg, const float* __restrict__ att,
                      const int* __restrict__ esrc, const int* __restrict__ edst,
                      const int* __restrict__ enode,
                      float* __restrict__ score, unsigned* __restrict__ mkey)
{
    int le = blockIdx.x * 8 + (threadIdx.x >> 5);
    if (le >= N_LE) return;
    int lane = threadIdx.x & 31;
    int src = esrc[le], dst = edst[le], node = enode[le];
    float acc = 0.0f;
    #pragma unroll
    for (int j = 0; j < 2; ++j) {
        int d = lane + 32 * j;
        float pre = xl[(size_t)src * D64 + d] + xr[(size_t)dst * D64 + d] +
                    eg[(size_t)node * D64 + d];
        pre = pre > 0.0f ? pre : 0.2f * pre;   // leaky 0.2
        acc += pre * att[d];
    }
    #pragma unroll
    for (int off = 16; off > 0; off >>= 1) acc += __shfl_xor(acc, off);
    if (lane == 0) {
        score[le] = acc;
        atomicMax(&mkey[dst], f2ord(acc));
    }
}

__global__ void gat_expsum_kernel(const float* sc, const unsigned* mkey,
                                  const int* edst, float* exb, float* ssum) {
    int le = blockIdx.x * blockDim.x + threadIdx.x;
    if (le >= N_LE) return;
    int dst = edst[le];
    float ex = expf(sc[le] - ord2f(mkey[dst]));
    exb[le] = ex;
    atomicAdd(&ssum[dst], ex);
}

__global__ __launch_bounds__(256)
void gat_out_kernel(const float* __restrict__ xl, const int* __restrict__ esrc,
                    const int* __restrict__ edst, const float* __restrict__ exb,
                    const float* __restrict__ ssum, float* __restrict__ out)
{
    int le = blockIdx.x * 8 + (threadIdx.x >> 5);
    if (le >= N_LE) return;
    int lane = threadIdx.x & 31;
    int src = esrc[le], dst = edst[le];
    float alpha = exb[le] / (ssum[dst] + 1e-16f);
    #pragma unroll
    for (int j = 0; j < 2; ++j) {
        int d = lane + 32 * j;
        atomicAdd(&out[(size_t)dst * D64 + d], alpha * xl[(size_t)src * D64 + d]);
    }
}

__global__ void gat_final_kernel(const float* outg, const float* b, float* out) {
    int i = blockIdx.x * blockDim.x + threadIdx.x;
    if (i < N_EDGES * D64) out[i] = outg[i] + b[i & 63];
}

// ---------------------------------------------------------------------------
// Host-side helpers
// ---------------------------------------------------------------------------
static void gemm(const float* A, const float* W, const float* bias, float* C,
                 int M, int K, int N, int act, hipStream_t s) {
    dim3 grid((M + BM - 1) / BM, N / BN);
    wmma_gemm_kernel<<<grid, dim3(256), 0, s>>>(A, W, bias, C, M, K, N, act);
}

static void tconv(const float* hin, int din, const float* e,
                  const float* Wq, const float* bq, const float* Wk, const float* bk,
                  const float* Wv, const float* bv, const float* Wsk, const float* bsk,
                  const int* esrc, const int* edst, const int* enode,
                  float* qb, float* kb, float* vb, float* skp, float* attout,
                  float* score, float* exb, unsigned* mkey, float* ssum,
                  float* hout, hipStream_t s)
{
    // hin stays L2-resident (<=102MB vs 192MB L2) across the 4 projections
    gemm(hin, Wq,  bq,  qb,  N_EDGES, din, HD, 0, s);
    gemm(hin, Wk,  bk,  kb,  N_EDGES, din, HD, 0, s);
    gemm(hin, Wv,  bv,  vb,  N_EDGES, din, HD, 0, s);
    gemm(hin, Wsk, bsk, skp, N_EDGES, din, HD, 0, s);
    hipMemsetAsync(ssum,   0, (size_t)N_EDGES * HEADS * sizeof(float), s);
    hipMemsetAsync(attout, 0, (size_t)N_EDGES * HD * sizeof(float), s);
    init_u32_kernel<<<(N_EDGES * HEADS + 255) / 256, 256, 0, s>>>(
        mkey, NEG_INF_ORD, N_EDGES * HEADS);
    tc_score_kernel<<<(N_LE + 7) / 8, 256, 0, s>>>(qb, kb, e, esrc, edst, enode,
                                                   score, mkey);
    tc_expsum_kernel<<<(N_LE * HEADS + 255) / 256, 256, 0, s>>>(score, mkey,
                                                                edst, exb, ssum);
    tc_out_kernel<<<(N_LE + 7) / 8, 256, 0, s>>>(vb, e, esrc, edst, enode,
                                                 exb, ssum, attout);
    add_kernel<<<(N_EDGES * HD + 255) / 256, 256, 0, s>>>(attout, skp, hout,
                                                          N_EDGES * HD);
}

// ---------------------------------------------------------------------------
// Entry point.  Workspace requirement: ~1.05 GB (261M floats).
// ---------------------------------------------------------------------------
extern "C" void kernel_launch(void* const* d_in, const int* in_sizes, int n_in,
                              void* d_out, int out_size, void* d_ws, size_t ws_size,
                              hipStream_t stream)
{
    (void)in_sizes; (void)n_in; (void)out_size; (void)ws_size;

    // ---- inputs, in setup_inputs() dict order (params flattened in order) ----
    const float* x    = (const float*)d_in[0];          // [50000,64]
    const float* h0   = (const float*)d_in[1];          // edge_attr [100000,64]
    const int*   lei  = (const int*)d_in[2];            // [2,400000]
    const int*   lnod = (const int*)d_in[3];            // [400000]
    const int* esrc = lei;
    const int* edst = lei + N_LE;

    const float* gine_We = (const float*)d_in[4];
    const float* gine_be = (const float*)d_in[5];
    const float* gine_ep = (const float*)d_in[6];       // scalar (1 elem)
    const float* gine_W1 = (const float*)d_in[7];
    const float* gine_b1 = (const float*)d_in[8];
    const float* gine_W2 = (const float*)d_in[9];
    const float* gine_b2 = (const float*)d_in[10];
    const float* gat_Wl  = (const float*)d_in[11];
    const float* gat_bl  = (const float*)d_in[12];
    const float* gat_Wr  = (const float*)d_in[13];
    const float* gat_br  = (const float*)d_in[14];
    const float* gat_We  = (const float*)d_in[15];
    const float* gat_att = (const float*)d_in[16];
    const float* gat_b   = (const float*)d_in[17];
    const float* t2_Wq = (const float*)d_in[18]; const float* t2_bq = (const float*)d_in[19];
    const float* t2_Wk = (const float*)d_in[20]; const float* t2_bk = (const float*)d_in[21];
    const float* t2_Wv = (const float*)d_in[22]; const float* t2_bv = (const float*)d_in[23];
    const float* t2_We = (const float*)d_in[24]; const float* t2_be = (const float*)d_in[25];
    const float* t2_Ws = (const float*)d_in[26]; const float* t2_bs = (const float*)d_in[27];
    const float* t3_Wq = (const float*)d_in[28]; const float* t3_bq = (const float*)d_in[29];
    const float* t3_Wk = (const float*)d_in[30]; const float* t3_bk = (const float*)d_in[31];
    const float* t3_Wv = (const float*)d_in[32]; const float* t3_bv = (const float*)d_in[33];
    const float* t3_We = (const float*)d_in[34]; const float* t3_be = (const float*)d_in[35];
    const float* t3_Ws = (const float*)d_in[36]; const float* t3_bs = (const float*)d_in[37];

    // ---- workspace bump allocator (float units) ----
    float* ws = (float*)d_ws;
    size_t off = 0;
    auto alloc = [&](size_t n) { float* p = ws + off; off += n; return p; };

    float* px_gine = alloc((size_t)N_NODES * D64);   // x @ gine_We + be  (hoisted)
    float* px_t2e  = alloc((size_t)N_NODES * HD);    // x @ tc2_We  + be
    float* px_t3e  = alloc((size_t)N_NODES * HD);    // x @ tc3_We  + be
    float* px_gat  = alloc((size_t)N_NODES * D64);   // x @ gat_We
    float* agg     = alloc((size_t)N_EDGES * D64);
    float* gbuf    = alloc((size_t)N_EDGES * D64);
    float* t1      = alloc((size_t)N_EDGES * D64);
    float* h64     = alloc((size_t)N_EDGES * D64);
    float* qb      = alloc((size_t)N_EDGES * HD);
    float* kb      = alloc((size_t)N_EDGES * HD);
    float* vb      = alloc((size_t)N_EDGES * HD);
    float* skp     = alloc((size_t)N_EDGES * HD);
    float* attout  = alloc((size_t)N_EDGES * HD);
    float* h256a   = alloc((size_t)N_EDGES * HD);
    float* h256b   = alloc((size_t)N_EDGES * HD);
    float* score   = alloc((size_t)N_LE * HEADS);
    float* exb     = alloc((size_t)N_LE * HEADS);
    unsigned* mkey = (unsigned*)alloc((size_t)N_EDGES * HEADS);
    float* ssum    = alloc((size_t)N_EDGES * HEADS);
    float* xl      = alloc((size_t)N_EDGES * D64);
    float* xr      = alloc((size_t)N_EDGES * D64);
    float* gsc     = alloc((size_t)N_LE);
    float* gex     = alloc((size_t)N_LE);
    unsigned* gm   = (unsigned*)alloc((size_t)N_EDGES);
    float* gs      = alloc((size_t)N_EDGES);
    float* outg    = alloc((size_t)N_EDGES * D64);

    // ---- Phase 0: hoisted node projections (50k rows instead of 400k) ----
    gemm(x, gine_We, gine_be, px_gine, N_NODES, D64, D64, 0, stream);
    gemm(x, t2_We,   t2_be,   px_t2e,  N_NODES, D64, HD,  0, stream);
    gemm(x, t3_We,   t3_be,   px_t3e,  N_NODES, D64, HD,  0, stream);
    gemm(x, gat_We,  nullptr, px_gat,  N_NODES, D64, D64, 0, stream);

    // ---- Phase 1: GINEConv ----
    hipMemsetAsync(agg, 0, (size_t)N_EDGES * D64 * sizeof(float), stream);
    gine_edge_kernel<<<(N_LE + 7) / 8, 256, 0, stream>>>(h0, px_gine, esrc, edst,
                                                         lnod, agg);
    gine_combine_kernel<<<(N_EDGES * D64 + 255) / 256, 256, 0, stream>>>(
        h0, agg, gine_ep, gbuf);
    gemm(gbuf, gine_W1, gine_b1, t1,  N_EDGES, D64, D64, 1, stream); // leaky .01
    gemm(t1,   gine_W2, gine_b2, h64, N_EDGES, D64, D64, 0, stream);

    // ---- Phase 2: TransformerConv x2 ----
    tconv(h64,   D64, px_t2e, t2_Wq, t2_bq, t2_Wk, t2_bk, t2_Wv, t2_bv,
          t2_Ws, t2_bs, esrc, edst, lnod, qb, kb, vb, skp, attout,
          score, exb, mkey, ssum, h256a, stream);
    tconv(h256a, HD,  px_t3e, t3_Wq, t3_bq, t3_Wk, t3_bk, t3_Wv, t3_bv,
          t3_Ws, t3_bs, esrc, edst, lnod, qb, kb, vb, skp, attout,
          score, exb, mkey, ssum, h256b, stream);

    // ---- Phase 3: GATv2Conv (heads=1) ----
    gemm(h256b, gat_Wl, gat_bl, xl, N_EDGES, HD, D64, 0, stream);
    gemm(h256b, gat_Wr, gat_br, xr, N_EDGES, HD, D64, 0, stream);
    hipMemsetAsync(gs,   0, (size_t)N_EDGES * sizeof(float), stream);
    hipMemsetAsync(outg, 0, (size_t)N_EDGES * D64 * sizeof(float), stream);
    init_u32_kernel<<<(N_EDGES + 255) / 256, 256, 0, stream>>>(gm, NEG_INF_ORD,
                                                               N_EDGES);
    gat_score_kernel<<<(N_LE + 7) / 8, 256, 0, stream>>>(xl, xr, px_gat, gat_att,
                                                         esrc, edst, lnod, gsc, gm);
    gat_expsum_kernel<<<(N_LE + 255) / 256, 256, 0, stream>>>(gsc, gm, edst,
                                                              gex, gs);
    gat_out_kernel<<<(N_LE + 7) / 8, 256, 0, stream>>>(xl, esrc, edst, gex, gs,
                                                       outg);
    gat_final_kernel<<<(N_EDGES * D64 + 255) / 256, 256, 0, stream>>>(
        outg, gat_b, (float*)d_out);
}
